// DotProductAttention_58901181497447
// MI455X (gfx1250) — matrix operand order repaired and checked by reading, not compiled, not run on hardware
//
#include <hip/hip_runtime.h>
#include <hip/hip_bf16.h>
#include <cstdint>
#include <cstddef>

#define NUM_NODES 100000
#define NUM_FEATS 128
#define NUM_EDGES 3200000
#define NUM_HEADS 4
#define QK_COLS   (2 * NUM_HEADS * NUM_FEATS)   // 1024

typedef __attribute__((ext_vector_type(16))) __bf16 v16bf;
typedef __attribute__((ext_vector_type(8)))  __bf16 v8bf;
typedef __attribute__((ext_vector_type(8)))  float  v8f;

union V16U { v16bf v; v8bf h[2]; };

// ---------------------------------------------------------------------------
// Split f32 -> (bf16 hi, bf16 lo) so that hi+lo ~= src to ~2^-16 relative.
// ---------------------------------------------------------------------------
__global__ void split_bf16_kernel(const float* __restrict__ src,
                                  __bf16* __restrict__ hi,
                                  __bf16* __restrict__ lo, int n) {
    int i = blockIdx.x * blockDim.x + threadIdx.x;
    if (i >= n) return;
    float v = src[i];
    __bf16 h = (__bf16)v;
    hi[i] = h;
    lo[i] = (__bf16)(v - (float)h);
}

__global__ void init_seg_kernel(float* __restrict__ seg_max,
                                float* __restrict__ seg_sum, int n) {
    int i = blockIdx.x * blockDim.x + threadIdx.x;
    if (i >= n) return;
    seg_max[i] = __int_as_float(0xFF800000);  // -inf
    seg_sum[i] = 0.0f;
}

// ---------------------------------------------------------------------------
// qk = x @ W^T + b via split-bf16 WMMA (f32 accumulate).
// Each wave computes one 16x16 tile; block = 8 waves = 16 rows x 128 cols.
// grid = (8, 6250): 8*8=64 N-tiles, 6250 M-tiles (6250*16 = 100000 exactly).
// ---------------------------------------------------------------------------
__global__ __launch_bounds__(256)
void gemm_qk_wmma(const __bf16* __restrict__ x_hi, const __bf16* __restrict__ x_lo,
                  const __bf16* __restrict__ w_hi, const __bf16* __restrict__ w_lo,
                  const float* __restrict__ bias, float* __restrict__ qk) {
    const int lane  = threadIdx.x & 31;
    const int wave  = threadIdx.x >> 5;
    const int mBase = blockIdx.y << 4;
    const int jBase = ((blockIdx.x << 3) + wave) << 4;
    const int hf    = lane >> 4;     // lane half (0/1)
    const int mr    = lane & 15;     // A: row within tile; B: col within tile

    const __bf16* xh = x_hi + (size_t)(mBase + mr) * NUM_FEATS;
    const __bf16* xl = x_lo + (size_t)(mBase + mr) * NUM_FEATS;
    const __bf16* wh = w_hi + (size_t)(jBase + mr) * NUM_FEATS;
    const __bf16* wl = w_lo + (size_t)(jBase + mr) * NUM_FEATS;

    v8f c = {};
#pragma unroll
    for (int kc = 0; kc < NUM_FEATS; kc += 32) {
        V16U ah, al, bh, bl;
        // A 16x32 bf16: lane (hf,mr): K = kc+hf*8+0..7 and kc+16+hf*8+0..7
        const int a0 = kc + hf * 8;
        ah.h[0] = *(const v8bf*)(xh + a0);
        ah.h[1] = *(const v8bf*)(xh + a0 + 16);
        al.h[0] = *(const v8bf*)(xl + a0);
        al.h[1] = *(const v8bf*)(xl + a0 + 16);
        // B 32x16 bf16: lane (hf,n=mr): K = kc+hf*8+0..7 and kc+16+hf*8+0..7 of W row (jBase+n)
        const int b0 = kc + hf * 8;
        bh.h[0] = *(const v8bf*)(wh + b0);
        bh.h[1] = *(const v8bf*)(wh + b0 + 16);
        bl.h[0] = *(const v8bf*)(wl + b0);
        bl.h[1] = *(const v8bf*)(wl + b0 + 16);
        // D = A_hi*B_hi + A_hi*B_lo + A_lo*B_hi + C  (drop lo*lo term)
        c = __builtin_amdgcn_wmma_f32_16x16x32_bf16(false, ah.v, false, bh.v,
                                                    (short)0, c, false, false);
        c = __builtin_amdgcn_wmma_f32_16x16x32_bf16(false, ah.v, false, bl.v,
                                                    (short)0, c, false, false);
        c = __builtin_amdgcn_wmma_f32_16x16x32_bf16(false, al.v, false, bh.v,
                                                    (short)0, c, false, false);
    }

    // C/D layout: VGPR r -> M = hf*8 + r, N = mr
    const int   col = jBase + mr;
    const float bv  = bias[col];
    float* outp = qk + (size_t)(mBase + hf * 8) * QK_COLS + col;
#pragma unroll
    for (int r = 0; r < 8; ++r)
        outp[(size_t)r * QK_COLS] = c[r] + bv;
}

// ---------------------------------------------------------------------------
// One wave per edge: scores[e,h] = q[row,h,:] . k[col,h,:], atomic seg-max.
// Lane l: head h = l>>3, sub = l&7 handles 16 feats via 4x float4.
// ---------------------------------------------------------------------------
__device__ __forceinline__ void atomic_max_f32(float* addr, float v) {
    if (v >= 0.0f) atomicMax((int*)addr, __float_as_int(v));
    else           atomicMin((unsigned int*)addr, __float_as_uint(v));
}

__global__ __launch_bounds__(256)
void edge_scores_kernel(const float* __restrict__ qk,
                        const long long* __restrict__ ei,
                        float* __restrict__ scores,
                        float* __restrict__ seg_max) {
    const int lane = threadIdx.x & 31;
    const int e    = blockIdx.x * 8 + (threadIdx.x >> 5);
    if (e >= NUM_EDGES) return;
    const int row = (int)ei[e];
    const int col = (int)ei[(size_t)NUM_EDGES + e];
    const int h   = lane >> 3;
    const int sub = lane & 7;

    const float4* qp =
        (const float4*)(qk + (size_t)row * QK_COLS + h * 2 * NUM_FEATS) + sub * 4;
    const float4* kp =
        (const float4*)(qk + (size_t)col * QK_COLS + h * 2 * NUM_FEATS + NUM_FEATS) + sub * 4;

    float s = 0.0f;
#pragma unroll
    for (int i = 0; i < 4; ++i) {
        float4 a = qp[i], b = kp[i];
        s += a.x * b.x + a.y * b.y + a.z * b.z + a.w * b.w;
    }
    s += __shfl_xor(s, 1, 32);
    s += __shfl_xor(s, 2, 32);
    s += __shfl_xor(s, 4, 32);
    if (sub == 0) {
        scores[(size_t)e * NUM_HEADS + h] = s;
        atomic_max_f32(seg_max + (size_t)row * NUM_HEADS + h, s);
    }
}

// ---------------------------------------------------------------------------
// ex = exp(score - seg_max[row]); in-place; atomic seg-sum.
// ---------------------------------------------------------------------------
__global__ void edge_expsum_kernel(const long long* __restrict__ ei,
                                   float* __restrict__ scores,
                                   const float* __restrict__ seg_max,
                                   float* __restrict__ seg_sum) {
    long long tid = (long long)blockIdx.x * blockDim.x + threadIdx.x;
    if (tid >= (long long)NUM_EDGES * NUM_HEADS) return;
    int e = (int)(tid >> 2), h = (int)(tid & 3);
    int row = (int)ei[e];
    float ex = __expf(scores[tid] - seg_max[(size_t)row * NUM_HEADS + h]);
    scores[tid] = ex;
    atomicAdd(seg_sum + (size_t)row * NUM_HEADS + h, ex);
}

// ---------------------------------------------------------------------------
// out[e] = mean_h( ex[e,h] / seg_sum[row,h] )
// ---------------------------------------------------------------------------
__global__ void edge_final_kernel(const long long* __restrict__ ei,
                                  const float* __restrict__ scores,
                                  const float* __restrict__ seg_sum,
                                  float* __restrict__ out) {
    int e = blockIdx.x * blockDim.x + threadIdx.x;
    if (e >= NUM_EDGES) return;
    int row = (int)ei[e];
    float4 ex = *(const float4*)(scores + (size_t)e * 4);
    float4 ss = *(const float4*)(seg_sum + (size_t)row * 4);
    out[e] = 0.25f * (ex.x / ss.x + ex.y / ss.y + ex.z / ss.z + ex.w / ss.w);
}

// ---------------------------------------------------------------------------
extern "C" void kernel_launch(void* const* d_in, const int* in_sizes, int n_in,
                              void* d_out, int out_size, void* d_ws, size_t ws_size,
                              hipStream_t stream) {
    const float*     x  = (const float*)d_in[0];      // 100000 x 128
    const float*     W  = (const float*)d_in[1];      // 1024 x 128
    const float*     b  = (const float*)d_in[2];      // 1024
    const long long* ei = (const long long*)d_in[3];  // 2 x 3200000 (int64)
    float* out = (float*)d_out;                       // 3200000

    // Workspace layout
    char* ws = (char*)d_ws;
    const size_t QK_BYTES  = (size_t)NUM_NODES * QK_COLS * sizeof(float);      // 409.6 MB
    const size_t SC_BYTES  = (size_t)NUM_EDGES * NUM_HEADS * sizeof(float);    // 51.2 MB
    const size_t SEG_BYTES = (size_t)NUM_NODES * NUM_HEADS * sizeof(float);    // 1.6 MB
    const size_t XH_BYTES  = (size_t)NUM_NODES * NUM_FEATS * sizeof(__bf16);   // 25.6 MB
    const size_t WH_BYTES  = (size_t)QK_COLS * NUM_FEATS * sizeof(__bf16);     // 256 KB

    size_t off = 0;
    float*  qk      = (float*)(ws + off); off += QK_BYTES;
    float*  scores  = (float*)(ws + off); off += SC_BYTES;
    float*  seg_max = (float*)(ws + off); off += SEG_BYTES;
    float*  seg_sum = (float*)(ws + off); off += SEG_BYTES;
    __bf16* x_hi    = (__bf16*)(ws + off); off += XH_BYTES;
    __bf16* x_lo    = (__bf16*)(ws + off); off += XH_BYTES;
    __bf16* w_hi    = (__bf16*)(ws + off); off += WH_BYTES;
    __bf16* w_lo    = (__bf16*)(ws + off); off += WH_BYTES;

    const int nx = NUM_NODES * NUM_FEATS;   // 12.8M
    const int nw = QK_COLS * NUM_FEATS;     // 131072
    const int ns = NUM_NODES * NUM_HEADS;   // 400K

    split_bf16_kernel<<<(nx + 255) / 256, 256, 0, stream>>>(x, x_hi, x_lo, nx);
    split_bf16_kernel<<<(nw + 255) / 256, 256, 0, stream>>>(W, w_hi, w_lo, nw);
    init_seg_kernel<<<(ns + 255) / 256, 256, 0, stream>>>(seg_max, seg_sum, ns);

    dim3 ggrid(QK_COLS / 128, NUM_NODES / 16);  // (8, 6250)
    gemm_qk_wmma<<<ggrid, 256, 0, stream>>>(x_hi, x_lo, w_hi, w_lo, b, qk);

    edge_scores_kernel<<<NUM_EDGES / 8, 256, 0, stream>>>(qk, ei, scores, seg_max);

    long long neh = (long long)NUM_EDGES * NUM_HEADS;  // 12.8M
    edge_expsum_kernel<<<(unsigned)((neh + 255) / 256), 256, 0, stream>>>(
        ei, scores, seg_max, seg_sum);

    edge_final_kernel<<<(NUM_EDGES + 255) / 256, 256, 0, stream>>>(
        ei, scores, seg_sum, out);
}
